// RelationalGraphConv_7645041787183
// MI455X (gfx1250) — matrix operand order repaired and checked by reading, not compiled, not run on hardware
//
#include <hip/hip_runtime.h>

#define D 64            // feature dim (D_IN == D_OUT)

typedef __attribute__((ext_vector_type(16))) __bf16 v16bf;
typedef __attribute__((ext_vector_type(8)))  float  v8f;

// ---------------------------------------------------------------------------
// Kernel 1: zero deg + update with vectorized grid-stride fill
// ---------------------------------------------------------------------------
__global__ void zero_ws_kernel(float4* p, long n4) {
    long i = (long)blockIdx.x * blockDim.x + threadIdx.x;
    long stride = (long)gridDim.x * blockDim.x;
    float4 z = make_float4(0.f, 0.f, 0.f, 0.f);
    for (; i < n4; i += stride) p[i] = z;
}

// ---------------------------------------------------------------------------
// Kernel 2: seg[e] = dst*R+rel; deg[seg] += w  (native no-return f32 fadd)
// ---------------------------------------------------------------------------
__global__ void deg_seg_kernel(const int* __restrict__ edst, const int* __restrict__ erel,
                               const float* __restrict__ ew, float* __restrict__ deg,
                               int* __restrict__ segA, int E, int R) {
    int e = blockIdx.x * blockDim.x + threadIdx.x;
    if (e >= E) return;
    int seg = edst[e] * R + erel[e];
    segA[e] = seg;
    __hip_atomic_fetch_add(&deg[seg], ew[e], __ATOMIC_RELAXED, __HIP_MEMORY_SCOPE_AGENT);
}

// ---------------------------------------------------------------------------
// Kernel 3: ewn[e] = w[e] / deg[seg[e]]  (one divide per edge, not per dim)
// ---------------------------------------------------------------------------
__global__ void ewn_kernel(const float* __restrict__ ew, const float* __restrict__ deg,
                           const int* __restrict__ segA, float* __restrict__ ewn, int E) {
    int e = blockIdx.x * blockDim.x + threadIdx.x;
    if (e >= E) return;
    ewn[e] = ew[e] / deg[segA[e]];
}

// ---------------------------------------------------------------------------
// Kernel 4: pre-pack [W_linear ; W_self] into the exact WMMA B register
// layout, bf16.  Layout: Bp[((chunk*4 + colTile)*32 + lane)*16 + j]
//   n = colTile*16 + (lane&15), khalf = lane>>4, kk = chunk*32 + (j<8 ? khalf*8 + j : 16 + khalf*8 + j-8)
// chunks 0..(RD/32-1) -> W_linear, last 2 chunks -> W_self
// ---------------------------------------------------------------------------
__global__ void packB_kernel(const float* __restrict__ Wl, const float* __restrict__ Ws,
                             __bf16* __restrict__ Bp, int RD, int total) {
    int t = blockIdx.x * blockDim.x + threadIdx.x;
    if (t >= total) return;
    int j  = t & 15;
    int L  = (t >> 4) & 31;
    int ct = (t >> 9) & 3;
    int c  = t >> 11;
    int n     = ct * 16 + (L & 15);
    int khalf = L >> 4;
    int kk    = c * 32 + ((j < 8) ? (khalf * 8 + j) : (16 + khalf * 8 + (j - 8)));
    float v = (kk < RD) ? Wl[(size_t)kk * D + n] : Ws[(size_t)(kk - RD) * D + n];
    Bp[t] = (__bf16)v;
}

// ---------------------------------------------------------------------------
// Kernel 5: scatter  update[seg, d] += x[src, d] * ewn   (64 threads / edge)
// ---------------------------------------------------------------------------
__global__ void scatter_kernel(const float* __restrict__ x,
                               const int* __restrict__ esrc,
                               const int* __restrict__ segA,
                               const float* __restrict__ ewn,
                               float* __restrict__ update, int E) {
    long gid = (long)blockIdx.x * blockDim.x + threadIdx.x;
    int e = (int)(gid >> 6);     // D == 64
    int d = (int)(gid & 63);
    if (e >= E) return;
    float v = x[(size_t)esrc[e] * D + d] * ewn[e];
    __hip_atomic_fetch_add(&update[(size_t)segA[e] * D + d], v,
                           __ATOMIC_RELAXED, __HIP_MEMORY_SCOPE_AGENT);
}

// ---------------------------------------------------------------------------
// Kernel 6: fused GEMM  out = relu(update @ Wl + x @ Ws + bl + bs)
// one wave -> 16x64 output strip: A fragment loaded ONCE per chunk, then
// 4 back-to-back v_wmma_f32_16x16x32_bf16 into 4 independent accumulators
// (no D->A/B RAW between them, hazard NOPs hidden). A traffic = 1x (102MB).
// ---------------------------------------------------------------------------
__device__ inline __bf16 cvt_bf16(float f) { return (__bf16)f; }

__device__ inline void pack_a8(v16bf& A, int off, float4 a, float4 b) {
    A[off + 0] = cvt_bf16(a.x); A[off + 1] = cvt_bf16(a.y);
    A[off + 2] = cvt_bf16(a.z); A[off + 3] = cvt_bf16(a.w);
    A[off + 4] = cvt_bf16(b.x); A[off + 5] = cvt_bf16(b.y);
    A[off + 6] = cvt_bf16(b.z); A[off + 7] = cvt_bf16(b.w);
}

__global__ void __launch_bounds__(256)
gemm_relu_kernel(const float* __restrict__ update,   // [N, RD] f32
                 const float* __restrict__ x,        // [N, D]  f32
                 const __bf16* __restrict__ Bp,      // packed B fragments
                 const float* __restrict__ bl,       // [D]
                 const float* __restrict__ bs,       // [D]
                 float* __restrict__ out,            // [N, D]
                 int N, int RD) {
    const int tid     = threadIdx.x;
    const int wave    = tid >> 5;
    const int lane    = tid & 31;
    const int rowBase = blockIdx.x * 128 + wave * 16;   // 16-row strip per wave
    const int mloc    = lane & 15;
    const int khalf   = lane >> 4;                      // 0 or 1

    int m      = rowBase + mloc;
    int mclamp = (m < N) ? m : (N - 1);                 // clamped loads: EXEC stays all-1s

    v8f c0, c1, c2, c3;
    #pragma unroll
    for (int i = 0; i < 8; ++i) { c0[i] = 0.f; c1[i] = 0.f; c2[i] = 0.f; c3[i] = 0.f; }

    const int nchunks = RD >> 5;                        // 16
    // per-lane B pointer: ((chunk*4 + colTile)*32 + lane)*16 bf16
    const __bf16* bpLane = Bp + (size_t)lane * 16;
    const size_t  bTile  = 32 * 16;                     // per colTile stride (bf16)
    const size_t  bChunk = 4 * 32 * 16;                 // per chunk stride (bf16)

    // ---- K-loop over update (RD = 512) in chunks of 32 ----
    const float* arow = update + (size_t)mclamp * RD;
    for (int kc = 0; kc < nchunks; ++kc) {
        const float4* pa = (const float4*)(arow + kc * 32 + khalf * 8);
        __builtin_prefetch(arow + (kc + 1) * 32 + khalf * 8, 0, 3);
        float4 a0 = pa[0], a1 = pa[1];                  // K = base .. base+7
        float4 a2 = pa[4], a3 = pa[5];                  // K = base+16 .. base+23
        v16bf A;
        pack_a8(A, 0, a0, a1);
        pack_a8(A, 8, a2, a3);
        const __bf16* bp = bpLane + (size_t)kc * bChunk;
        v16bf B0 = *(const v16bf*)(bp + 0 * bTile);
        v16bf B1 = *(const v16bf*)(bp + 1 * bTile);
        v16bf B2 = *(const v16bf*)(bp + 2 * bTile);
        v16bf B3 = *(const v16bf*)(bp + 3 * bTile);
        c0 = __builtin_amdgcn_wmma_f32_16x16x32_bf16(false, A, false, B0, (short)0, c0, false, false);
        c1 = __builtin_amdgcn_wmma_f32_16x16x32_bf16(false, A, false, B1, (short)0, c1, false, false);
        c2 = __builtin_amdgcn_wmma_f32_16x16x32_bf16(false, A, false, B2, (short)0, c2, false, false);
        c3 = __builtin_amdgcn_wmma_f32_16x16x32_bf16(false, A, false, B3, (short)0, c3, false, false);
    }

    // ---- self-loop: K = 64 over x, chunks nchunks..nchunks+1 of packed B ----
    const float* xrow = x + (size_t)mclamp * D;
    #pragma unroll
    for (int kc = 0; kc < 2; ++kc) {
        const float4* pa = (const float4*)(xrow + kc * 32 + khalf * 8);
        float4 a0 = pa[0], a1 = pa[1];
        float4 a2 = pa[4], a3 = pa[5];
        v16bf A;
        pack_a8(A, 0, a0, a1);
        pack_a8(A, 8, a2, a3);
        const __bf16* bp = bpLane + (size_t)(nchunks + kc) * bChunk;
        v16bf B0 = *(const v16bf*)(bp + 0 * bTile);
        v16bf B1 = *(const v16bf*)(bp + 1 * bTile);
        v16bf B2 = *(const v16bf*)(bp + 2 * bTile);
        v16bf B3 = *(const v16bf*)(bp + 3 * bTile);
        c0 = __builtin_amdgcn_wmma_f32_16x16x32_bf16(false, A, false, B0, (short)0, c0, false, false);
        c1 = __builtin_amdgcn_wmma_f32_16x16x32_bf16(false, A, false, B1, (short)0, c1, false, false);
        c2 = __builtin_amdgcn_wmma_f32_16x16x32_bf16(false, A, false, B2, (short)0, c2, false, false);
        c3 = __builtin_amdgcn_wmma_f32_16x16x32_bf16(false, A, false, B3, (short)0, c3, false, false);
    }

    // ---- bias + ReLU + store
    // C layout per tile t: row = rowBase + v + 8*khalf, col = t*16 + mloc ----
    #pragma unroll
    for (int t = 0; t < 4; ++t) {
        const v8f& c = (t == 0) ? c0 : (t == 1) ? c1 : (t == 2) ? c2 : c3;
        int n = t * 16 + mloc;
        float bias = bl[n] + bs[n];
        #pragma unroll
        for (int v = 0; v < 8; ++v) {
            int mo = rowBase + v + khalf * 8;
            if (mo < N) {
                float val = c[v] + bias;
                out[(size_t)mo * D + n] = val > 0.f ? val : 0.f;
            }
        }
    }
}

// ---------------------------------------------------------------------------
extern "C" void kernel_launch(void* const* d_in, const int* in_sizes, int n_in,
                              void* d_out, int out_size, void* d_ws, size_t ws_size,
                              hipStream_t stream) {
    const float* x    = (const float*)d_in[0];
    const int*   esrc = (const int*)  d_in[1];
    const int*   edst = (const int*)  d_in[2];
    const int*   erel = (const int*)  d_in[3];
    const float* ew   = (const float*)d_in[4];
    const float* Wl   = (const float*)d_in[5];
    const float* bl   = (const float*)d_in[6];
    const float* Ws   = (const float*)d_in[7];
    const float* bs   = (const float*)d_in[8];
    float* out = (float*)d_out;

    const int N  = in_sizes[0] / D;              // 50000
    const int E  = in_sizes[1];                  // 1000000
    const int R  = in_sizes[5] / (D * D);        // 8
    const int RD = R * D;                        // 512

    // ---- workspace layout ----
    float* deg    = (float*)d_ws;                          // [N*R]
    float* update = deg + (size_t)N * R;                   // [N*R, D]
    float* ewn    = update + (size_t)N * R * D;            // [E]
    int*   segA   = (int*)(ewn + E);                       // [E]
    __bf16* Bp    = (__bf16*)(segA + E);                   // [(RD/32+2)*4*32*16]

    // zero deg + update (contiguous)
    long totalF = (long)N * R + (long)N * R * D;
    long n4 = (totalF + 3) / 4;
    zero_ws_kernel<<<2048, 256, 0, stream>>>((float4*)d_ws, n4);

    // degrees + segment ids
    deg_seg_kernel<<<(E + 255) / 256, 256, 0, stream>>>(edst, erel, ew, deg, segA, E, R);

    // normalized edge weights
    ewn_kernel<<<(E + 255) / 256, 256, 0, stream>>>(ew, deg, segA, ewn, E);

    // pre-pack weights into WMMA B fragment layout (bf16)
    const int nchunksB = RD / 32 + 2;            // 18
    const int totalB = nchunksB * 4 * 32 * 16;   // 36864
    packB_kernel<<<(totalB + 255) / 256, 256, 0, stream>>>(Wl, Ws, Bp, RD, totalB);

    // scatter-add messages
    long sthreads = (long)E * D;
    scatter_kernel<<<(unsigned)((sthreads + 255) / 256), 256, 0, stream>>>(
        x, esrc, segA, ewn, update, E);

    // fused GEMM + self-loop + bias + relu (one wave per 16x64 strip)
    gemm_relu_kernel<<<(N + 127) / 128, 256, 0, stream>>>(
        update, x, Bp, bl, bs, out, N, RD);
}
